// GGNNLayer_42030549959140
// MI455X (gfx1250) — matrix operand order repaired and checked by reading, not compiled, not run on hardware
//
#include <hip/hip_runtime.h>
#include <hip/hip_bf16.h>

// ---------------- problem constants ----------------
#define N_NODES 50000
#define N_EDGES 800000
#define DIM     128
#define K2      256          // 2*DIM
#define NTILES  (N_NODES/16) // 3125, exact
#define LN_EPS  1e-5f
#define WPB     2            // waves (16-row tiles) per block

typedef __attribute__((ext_vector_type(16))) _Float16 v16h;
typedef __attribute__((ext_vector_type(8)))  float    v8f;
typedef _Float16 half_t;

// ---------------- helpers ----------------
__device__ __forceinline__ v8f wmma_f16(v16h a, v16h b, v8f c) {
    // D = A(16x32 f16) * B(32x16 f16) + C(16x16 f32)
    return __builtin_amdgcn_wmma_f32_16x16x32_f16(
        /*neg_a=*/false, a, /*neg_b=*/false, b,
        /*c_mod=*/(short)0, c, /*reuse_a=*/false, /*reuse_b=*/false);
}

// Build an A fragment (16x32, f16) from an LDS tile stored row-major [16][128] f16.
// ISA layout: lanes 0-15 hold M=lane, halves 0-7 = K k0..k0+7, halves 8-15 = K k0+16..k0+23
// with k0 = kk; lanes 16-31 same M mapping with k0 = kk+8.
__device__ __forceinline__ v16h load_a(const half_t* tile, int lane, int kk) {
    int m  = lane & 15;
    int k0 = kk + ((lane >> 4) << 3);
    const half_t* p = tile + m * DIM + k0;
    uint4 t0 = *(const uint4*)(p);        // halves k0 .. k0+7
    uint4 t1 = *(const uint4*)(p + 16);   // halves k0+16 .. k0+23
    v16h a;
    ((uint4*)&a)[0] = t0;
    ((uint4*)&a)[1] = t1;
    return a;
}

// B fragments are pre-packed per (matrix w, ntile t, kstep s, lane): 16 halves each.
__device__ __forceinline__ v16h load_b(const half_t* wf, int w, int t, int s, int lane) {
    const half_t* p = wf + (size_t)(((w << 6) + (t << 3) + s)) * 512 + lane * 16;
    return *(const v16h*)p;              // 32-byte aligned
}

// ---------------- kernel 0: zero scratch ----------------
__global__ void ggnn_zero(float* p, int n) {
    int i = blockIdx.x * blockDim.x + threadIdx.x;
    if (i < n) p[i] = 0.0f;
}

// ---------------- kernel 1: pack weights into WMMA B-fragment layout ----------------
// B(32x16) layout: lane<16 -> column n=lane,   halves h -> K = s*32 + (h<8 ? h : h+8)
//                  lane>=16 -> column n=lane-16, halves h -> K = s*32 + 8 + (h<8 ? h : h+8)
// B[k][n] = W[t*16+n][k]  (W row-major [128][256])
__global__ void ggnn_wprep(const float* __restrict__ Wz,
                           const float* __restrict__ Wr,
                           const float* __restrict__ Wh,
                           half_t* __restrict__ wf) {
    int gid = blockIdx.x * blockDim.x + threadIdx.x;
    if (gid >= 3 * 64 * 32) return;
    int L = gid & 31;
    int f = gid >> 5;            // 0..191
    int s = f & 7;               // kstep
    int t = (f >> 3) & 7;        // ntile
    int w = f >> 6;              // matrix id
    const float* W = (w == 0) ? Wz : ((w == 1) ? Wr : Wh);
    int n     = L & 15;
    int kbase = s * 32 + ((L >> 4) ? 8 : 0);
    const float* row = W + (size_t)(t * 16 + n) * K2;
    half_t* dst = wf + (size_t)f * 512 + L * 16;
#pragma unroll
    for (int h = 0; h < 16; ++h) dst[h] = (half_t)row[kbase + (h < 8 ? h : h + 8)];
}

// ---------------- kernel 2: edge aggregation (wave per edge) ----------------
__global__ void ggnn_aggregate(const float* __restrict__ x,
                               const int*   __restrict__ ei,
                               const float* __restrict__ ew,
                               float* __restrict__ msg,
                               float* __restrict__ deg) {
    int tid  = blockIdx.x * blockDim.x + threadIdx.x;
    int e    = tid >> 5;
    int lane = tid & 31;
    if (e >= N_EDGES) return;
    int   src = ei[e];
    int   dst = ei[N_EDGES + e];
    float w   = ew[e];
    float4 xv = *(const float4*)(x + (size_t)src * DIM + lane * 4);
    float* mp = msg + (size_t)dst * DIM + lane * 4;
    __hip_atomic_fetch_add(mp + 0, xv.x * w, __ATOMIC_RELAXED, __HIP_MEMORY_SCOPE_AGENT);
    __hip_atomic_fetch_add(mp + 1, xv.y * w, __ATOMIC_RELAXED, __HIP_MEMORY_SCOPE_AGENT);
    __hip_atomic_fetch_add(mp + 2, xv.z * w, __ATOMIC_RELAXED, __HIP_MEMORY_SCOPE_AGENT);
    __hip_atomic_fetch_add(mp + 3, xv.w * w, __ATOMIC_RELAXED, __HIP_MEMORY_SCOPE_AGENT);
    if (lane == 0)
        __hip_atomic_fetch_add(deg + dst, 1.0f, __ATOMIC_RELAXED, __HIP_MEMORY_SCOPE_AGENT);
}

// ---------------- kernel 3: fused GRU cell + LayerNorm (wave per 16-row tile) ----------------
__global__ void __launch_bounds__(WPB * 32)
ggnn_gru(const float* __restrict__ x,
         const float* __restrict__ msg,
         const float* __restrict__ deg,
         const half_t* __restrict__ wf,
         const float* __restrict__ b_z,
         const float* __restrict__ b_r,
         const float* __restrict__ b_h,
         const float* __restrict__ gamma,
         const float* __restrict__ beta,
         float* __restrict__ out) {
    __shared__ __align__(32) float  xf[WPB][16 * DIM];  // x fp32
    __shared__ __align__(32) float  zf[WPB][16 * DIM];  // z, then x_new
    __shared__ __align__(32) half_t xh[WPB][16 * DIM];  // x  f16
    __shared__ __align__(32) half_t mh[WPB][16 * DIM];  // m  f16 (normalized)
    __shared__ __align__(32) half_t ah[WPB][16 * DIM];  // r*x f16

    const int wid  = threadIdx.x >> 5;
    const int lane = threadIdx.x & 31;
    const int tile = blockIdx.x * WPB + wid;
    if (tile >= NTILES) return;               // wave-uniform; EXEC stays all-1 for WMMA

    // ---- stage tile: x (f32+f16), m = msg/clamp(deg,1) (f16) ----
    const float* xrow = x   + (size_t)tile * 16 * DIM;
    const float* mrow = msg + (size_t)tile * 16 * DIM;
#pragma unroll 4
    for (int i = lane; i < 512; i += 32) {    // 512 float4 chunks
        int r = i >> 5;
        int c = (i & 31) << 2;
        int o = r * DIM + c;
        float4 xv = *(const float4*)(xrow + o);
        float4 mv = *(const float4*)(mrow + o);
        float  dv = fmaxf(deg[tile * 16 + r], 1.0f);
        float inv = 1.0f / dv;
        *(float4*)&xf[wid][o] = xv;
        xh[wid][o + 0] = (half_t)xv.x; xh[wid][o + 1] = (half_t)xv.y;
        xh[wid][o + 2] = (half_t)xv.z; xh[wid][o + 3] = (half_t)xv.w;
        mh[wid][o + 0] = (half_t)(mv.x * inv); mh[wid][o + 1] = (half_t)(mv.y * inv);
        mh[wid][o + 2] = (half_t)(mv.z * inv); mh[wid][o + 3] = (half_t)(mv.w * inv);
    }

    const int col_lo = lane & 15;
    const int rbase  = (lane >> 4) << 3;      // C/D layout: lanes 0-15 rows 0-7, 16-31 rows 8-15

    // ---- pass 1: z = sigmoid([x,m] Wz^T + bz), r = sigmoid([x,m] Wr^T + br) ----
#pragma unroll 2
    for (int t = 0; t < 8; ++t) {
        int col = t * 16 + col_lo;
        float bzv = b_z[col], brv = b_r[col];
        v8f az, ar;
#pragma unroll
        for (int i = 0; i < 8; ++i) { az[i] = bzv; ar[i] = brv; }
#pragma unroll
        for (int s = 0; s < 8; ++s) {
            const half_t* abase = (s < 4) ? xh[wid] : mh[wid];
            int kk = (s < 4) ? s * 32 : (s - 4) * 32;
            v16h a = load_a(abase, lane, kk);
            az = wmma_f16(a, load_b(wf, 0, t, s, lane), az);
            ar = wmma_f16(a, load_b(wf, 1, t, s, lane), ar);
        }
#pragma unroll
        for (int i = 0; i < 8; ++i) {
            int row = rbase + i;
            int o   = row * DIM + col;
            float z = 1.0f / (1.0f + __expf(-az[i]));
            float r = 1.0f / (1.0f + __expf(-ar[i]));
            float xv = xf[wid][o];
            zf[wid][o] = z;
            ah[wid][o] = (half_t)(r * xv);
        }
    }

    // ---- pass 2: h~ = tanh([r*x, m] Wh^T + bh); x_new = (1-z)x + z h~ ----
#pragma unroll 2
    for (int t = 0; t < 8; ++t) {
        int col = t * 16 + col_lo;
        float bhv = b_h[col];
        v8f ac;
#pragma unroll
        for (int i = 0; i < 8; ++i) ac[i] = bhv;
#pragma unroll
        for (int s = 0; s < 8; ++s) {
            const half_t* abase = (s < 4) ? ah[wid] : mh[wid];
            int kk = (s < 4) ? s * 32 : (s - 4) * 32;
            v16h a = load_a(abase, lane, kk);
            ac = wmma_f16(a, load_b(wf, 2, t, s, lane), ac);
        }
#pragma unroll
        for (int i = 0; i < 8; ++i) {
            int row = rbase + i;
            int o   = row * DIM + col;
            float ht = tanhf(ac[i]);
            float z  = zf[wid][o];
            float xv = xf[wid][o];
            zf[wid][o] = (1.0f - z) * xv + z * ht;   // overwrite z with x_new (same slot)
        }
    }

    // ---- LayerNorm per row + store (wave-shuffle reduction, 4 cols/lane) ----
    const int c0 = lane * 4;
    for (int row = 0; row < 16; ++row) {
        float4 v = *(const float4*)&zf[wid][row * DIM + c0];
        float s = v.x + v.y + v.z + v.w;
#pragma unroll
        for (int off = 16; off >= 1; off >>= 1) s += __shfl_xor(s, off, 32);
        float mu = s * (1.0f / 128.0f);
        float dx0 = v.x - mu, dx1 = v.y - mu, dx2 = v.z - mu, dx3 = v.w - mu;
        float q = dx0 * dx0 + dx1 * dx1 + dx2 * dx2 + dx3 * dx3;
#pragma unroll
        for (int off = 16; off >= 1; off >>= 1) q += __shfl_xor(q, off, 32);
        float inv = rsqrtf(q * (1.0f / 128.0f) + LN_EPS);
        float4 o;
        o.x = dx0 * inv * gamma[c0 + 0] + beta[c0 + 0];
        o.y = dx1 * inv * gamma[c0 + 1] + beta[c0 + 1];
        o.z = dx2 * inv * gamma[c0 + 2] + beta[c0 + 2];
        o.w = dx3 * inv * gamma[c0 + 3] + beta[c0 + 3];
        *(float4*)(out + ((size_t)tile * 16 + row) * DIM + c0) = o;
    }
}

// ---------------- host launch ----------------
extern "C" void kernel_launch(void* const* d_in, const int* in_sizes, int n_in,
                              void* d_out, int out_size, void* d_ws, size_t ws_size,
                              hipStream_t stream) {
    const float* x  = (const float*)d_in[0];
    const int*   ei = (const int*)  d_in[1];
    const float* ew = (const float*)d_in[2];
    const float* Wz = (const float*)d_in[3];
    const float* bz = (const float*)d_in[4];
    const float* Wr = (const float*)d_in[5];
    const float* br = (const float*)d_in[6];
    const float* Wh = (const float*)d_in[7];
    const float* bh = (const float*)d_in[8];
    const float* g  = (const float*)d_in[9];
    const float* be = (const float*)d_in[10];
    float* out = (float*)d_out;

    // workspace: messages [N*128] f32 | degree [N] f32 | weight frags [3*64*512] f16
    float*  msg = (float*)d_ws;
    float*  deg = msg + (size_t)N_NODES * DIM;
    half_t* wf  = (half_t*)(deg + N_NODES);   // byte offset 25,800,000: 32B aligned

    int zn = N_NODES * (DIM + 1);             // messages + degree, contiguous
    ggnn_zero<<<(zn + 255) / 256, 256, 0, stream>>>(msg, zn);
    ggnn_wprep<<<(3 * 64 * 32 + 255) / 256, 256, 0, stream>>>(Wz, Wr, Wh, wf);
    ggnn_aggregate<<<(N_EDGES * 32) / 256, 256, 0, stream>>>(x, ei, ew, msg, deg);
    ggnn_gru<<<(NTILES + WPB - 1) / WPB, WPB * 32, 0, stream>>>(
        x, msg, deg, wf, bz, br, bh, g, be, out);
}
